// Naive_CNN_46024869544366
// MI455X (gfx1250) — matrix operand order repaired and checked by reading, not compiled, not run on hardware
//
#include <hip/hip_runtime.h>

// Spiking CNN forward (T=8): 3x [conv3x3 -> train-BN -> LIF -> maxpool2], time-mean.
// conv1 computed once (input is T-invariant); conv2/conv3 are per-tap K=32 implicit
// GEMMs on {0,1} f16 spikes via V_WMMA_F32_16X16X32_F16. Spike buffers are halo-padded
// (zero border) so conv B-fragment loads are branchless b128 loads at immediate offsets.
// Conv biases are absorbed by train-mode BN and skipped. BN stats: deterministic 2-pass.

typedef __attribute__((ext_vector_type(16))) _Float16 v16h;
typedef __attribute__((ext_vector_type(8)))  float    v8f;

#define TT 8
#define BB 64

// p1: per image 34x34 positions (1-px zero halo) x 32ch f16, channel-last.
#define P1_ROW   34
#define P1_IMG   (34 * 34)          // 1156 positions
// p2: per image 18x18 positions x 32ch f16.
#define P2_ROW   18
#define P2_IMG   (18 * 18)          // 324 positions

// ------------------------------------------------------------------ zero (halo) fill
__global__ void zero_kernel(float4* __restrict__ p, int n16) {
  int i      = blockIdx.x * blockDim.x + threadIdx.x;
  int stride = gridDim.x * blockDim.x;
  float4 z = {0.f, 0.f, 0.f, 0.f};
  for (; i < n16; i += stride) p[i] = z;
}

// ---------------------------------------------------------------- conv1 (3->32, 64x64)
// x: (64,3,64,64) NCHW. y1: channel-last (b, pix=4096, c=32) f32. One t only (T-invariant).
__global__ void conv1_kernel(const float* __restrict__ x, const float* __restrict__ w1,
                             float* __restrict__ y1) {
  __shared__ float ws[32 * 27];
  for (int i = threadIdx.x; i < 32 * 27; i += blockDim.x) ws[i] = w1[i];
  __syncthreads();
  int tid = blockIdx.x * blockDim.x + threadIdx.x;   // 64*4096*32
  int c   = tid & 31;
  int pix = (tid >> 5) & 4095;
  int b   = tid >> 17;
  int py = pix >> 6, px = pix & 63;
  const float* xb = x + b * 3 * 4096;
  const float* wc = ws + c * 27;
  float acc = 0.f;
#pragma unroll
  for (int cin = 0; cin < 3; ++cin)
#pragma unroll
    for (int dy = 0; dy < 3; ++dy) {
      int yy = py + dy - 1;
#pragma unroll
      for (int dx = 0; dx < 3; ++dx) {
        int xx = px + dx - 1;
        float v = (yy >= 0 && yy < 64 && xx >= 0 && xx < 64) ? xb[cin * 4096 + yy * 64 + xx] : 0.f;
        acc = fmaf(v, wc[cin * 9 + dy * 3 + dx], acc);
      }
    }
  y1[tid] = acc;   // tid == ((b*4096)+pix)*32 + c  -> channel-last
}

// ------------------------------------------------- deterministic BN stats (channel-last src)
__global__ void stats_partial_kernel(const float* __restrict__ src, int C, int nrows,
                                     float* __restrict__ partial) {
  __shared__ float ssum[256], ssq[256];
  int c      = threadIdx.x % C;
  int sub    = threadIdx.x / C;
  int perBlk = 256 / C;
  int row    = blockIdx.x * perBlk + sub;
  int stride = gridDim.x * perBlk;
  float s = 0.f, q = 0.f;
  for (; row < nrows; row += stride) {
    float v = src[row * C + c];
    s += v; q += v * v;
  }
  ssum[threadIdx.x] = s; ssq[threadIdx.x] = q;
  __syncthreads();
  if (sub == 0) {
    for (int i = 1; i < perBlk; ++i) { s += ssum[i * C + c]; q += ssq[i * C + c]; }
    partial[blockIdx.x * 2 * C + c]     = s;
    partial[blockIdx.x * 2 * C + C + c] = q;
  }
}

// ss[c] = gamma*rsqrt(var+eps); ss[C+c] = beta - mean*scale
__global__ void finalize_stats_kernel(const float* __restrict__ partial, int nblk, int C,
                                      float invCnt, const float* __restrict__ gamma,
                                      const float* __restrict__ beta, float* __restrict__ ss) {
  int c = threadIdx.x;
  if (c >= C) return;
  float s = 0.f, q = 0.f;
  for (int i = 0; i < nblk; ++i) { s += partial[i * 2 * C + c]; q += partial[i * 2 * C + C + c]; }
  float mean  = s * invCnt;
  float var   = q * invCnt - mean * mean;
  float scale = gamma[c] * rsqrtf(var + 1e-5f);
  ss[c]     = scale;
  ss[C + c] = beta[c] - mean * scale;
}

// ------------------------------------------- fused BN + LIF(T=8, const input) + pool (block1)
// y1: (b,4096,32) f32 -> p1 interior: (t*64+b, 34x34 padded, 32) f16 spikes.
__global__ void bn_lif_pool1_kernel(const float* __restrict__ y1, const float* __restrict__ ss,
                                    _Float16* __restrict__ p1) {
  int tid  = blockIdx.x * blockDim.x + threadIdx.x;  // 64*1024*32
  int c    = tid & 31;
  int owin = (tid >> 5) & 1023;
  int b    = tid >> 15;
  int oy = owin >> 5, ox = owin & 31;
  float scale = ss[c], shift = ss[32 + c];
  int base = (b * 4096 + (2 * oy) * 64 + 2 * ox) * 32 + c;
  float x0 = fmaf(scale, y1[base],        shift);
  float x1 = fmaf(scale, y1[base + 32],   shift);
  float x2 = fmaf(scale, y1[base + 2048], shift);
  float x3 = fmaf(scale, y1[base + 2080], shift);
  float v0 = 0.f, v1 = 0.f, v2 = 0.f, v3 = 0.f;
  size_t opos = (size_t)(oy + 1) * P1_ROW + (ox + 1);
#pragma unroll
  for (int t = 0; t < TT; ++t) {
    v0 = 0.5f * (v0 + x0); v1 = 0.5f * (v1 + x1);
    v2 = 0.5f * (v2 + x2); v3 = 0.5f * (v3 + x3);
    float s0 = (v0 >= 1.f) ? 1.f : 0.f;  float s1 = (v1 >= 1.f) ? 1.f : 0.f;
    float s2 = (v2 >= 1.f) ? 1.f : 0.f;  float s3 = (v3 >= 1.f) ? 1.f : 0.f;
    v0 *= (1.f - s0); v1 *= (1.f - s1); v2 *= (1.f - s2); v3 *= (1.f - s3);
    float pool = fmaxf(fmaxf(s0, s1), fmaxf(s2, s3));
    p1[((size_t)(t * BB + b) * P1_IMG + opos) * 32 + c] = (_Float16)pool;
  }
}

// --------------------------------------------------- weight packing into WMMA A-fragments
// A-frag layout used here (contiguous k, same order as B loads): lane = m + 16*(k/16), half h = k%16.
// wf: [tap][mtile][lane][16h]
__global__ void pack_w_kernel(const float* __restrict__ w, _Float16* __restrict__ wf,
                              int nMtiles, int total) {
  int idx = blockIdx.x * blockDim.x + threadIdx.x;
  if (idx >= total) return;
  int h    = idx & 15;
  int lane = (idx >> 4) & 31;
  int rest = idx >> 9;               // mt + nMtiles*tap
  int mt   = rest % nMtiles;
  int tap  = rest / nMtiles;
  int m = lane & 15, side = lane >> 4;
  int k = 16 * side + h;   // cin
  int cout = mt * 16 + m;
  wf[idx] = (_Float16)w[(cout * 32 + k) * 9 + tap];
}

// --------------------------------------------------------------- conv2 WMMA (32->32, 32x32)
// p1 (halo-padded spikes) -> y2: (tb, 1024, 32) f32. One wave per 16-pixel row segment.
// All 9 tap loads are branchless: one per-lane base + immediate offsets.
__global__ void conv2_wmma_kernel(const _Float16* __restrict__ p1,
                                  const _Float16* __restrict__ wf2,
                                  float* __restrict__ y2) {
  int gtid = blockIdx.x * blockDim.x + threadIdx.x;
  int wave = gtid >> 5;              // 512 tb * 64 ntiles
  int lane = threadIdx.x & 31;
  int ntile = wave & 63;
  int tb    = wave >> 6;
  int y     = ntile >> 1;
  int xbase = (ntile & 1) << 4;
  int n       = lane & 15;
  int cinbase = (lane >> 4) << 4;
  v8f c0 = {}, c1 = {};
  // padded coords: tap (dy,dx) reads stored row y+dy, col xbase+n+dx
  const _Float16* bptr = p1 + (size_t)tb * (P1_IMG * 32)
                            + ((size_t)y * P1_ROW + xbase + n) * 32 + cinbase;
#pragma unroll
  for (int dy = 0; dy < 3; ++dy)
#pragma unroll
    for (int dx = 0; dx < 3; ++dx) {
      int tap = dy * 3 + dx;
      v16h bfrag = *(const v16h*)(bptr + (dy * P1_ROW + dx) * 32);
      v16h a0 = *(const v16h*)(wf2 + ((tap * 2 + 0) * 32 + lane) * 16);
      v16h a1 = *(const v16h*)(wf2 + ((tap * 2 + 1) * 32 + lane) * 16);
      c0 = __builtin_amdgcn_wmma_f32_16x16x32_f16(false, a0, false, bfrag, (short)0, c0, false, false);
      c1 = __builtin_amdgcn_wmma_f32_16x16x32_f16(false, a1, false, bfrag, (short)0, c1, false, false);
    }
  // D layout: vgpr r -> (m = r + 8*(lane>>4), npix = lane&15)
  int mAdd = (lane >> 4) * 8;
  size_t obase = ((size_t)tb * 1024 + y * 32 + xbase + n) * 32;
#pragma unroll
  for (int r = 0; r < 8; ++r) {
    y2[obase + mAdd + r]      = c0[r];
    y2[obase + 16 + mAdd + r] = c1[r];
  }
}

// ----------------------------------------------- fused BN + LIF + pool (block2)
// y2 (tb,1024,32) f32 -> p2 interior (tb, 18x18 padded, 32) f16 spikes.
__global__ void bn_lif_pool2_kernel(const float* __restrict__ y2, const float* __restrict__ ss,
                                    _Float16* __restrict__ p2) {
  int tid  = blockIdx.x * blockDim.x + threadIdx.x;  // 64*256*32
  int c    = tid & 31;
  int owin = (tid >> 5) & 255;
  int b    = tid >> 13;
  int oy = owin >> 4, ox = owin & 15;
  float scale = ss[c], shift = ss[32 + c];
  float v0 = 0.f, v1 = 0.f, v2 = 0.f, v3 = 0.f;
  size_t opos = (size_t)(oy + 1) * P2_ROW + (ox + 1);
#pragma unroll
  for (int t = 0; t < TT; ++t) {
    size_t base = ((size_t)(t * BB + b) * 1024 + (2 * oy) * 32 + 2 * ox) * 32 + c;
    float x0 = fmaf(scale, y2[base],        shift);
    float x1 = fmaf(scale, y2[base + 32],   shift);
    float x2 = fmaf(scale, y2[base + 1024], shift);
    float x3 = fmaf(scale, y2[base + 1056], shift);
    v0 = 0.5f * (v0 + x0); v1 = 0.5f * (v1 + x1);
    v2 = 0.5f * (v2 + x2); v3 = 0.5f * (v3 + x3);
    float s0 = (v0 >= 1.f) ? 1.f : 0.f;  float s1 = (v1 >= 1.f) ? 1.f : 0.f;
    float s2 = (v2 >= 1.f) ? 1.f : 0.f;  float s3 = (v3 >= 1.f) ? 1.f : 0.f;
    v0 *= (1.f - s0); v1 *= (1.f - s1); v2 *= (1.f - s2); v3 *= (1.f - s3);
    p2[((size_t)(t * BB + b) * P2_IMG + opos) * 32 + c]
        = (_Float16)fmaxf(fmaxf(s0, s1), fmaxf(s2, s3));
  }
}

// --------------------------------------------------------------- conv3 WMMA (32->64, 16x16)
// p2 (halo-padded) -> y3: (tb, 256, 64) f32. One wave per 16-pixel row, 4 cout tiles.
__global__ void conv3_wmma_kernel(const _Float16* __restrict__ p2,
                                  const _Float16* __restrict__ wf3,
                                  float* __restrict__ y3) {
  int gtid = blockIdx.x * blockDim.x + threadIdx.x;
  int wave = gtid >> 5;              // 512 tb * 16 rows
  int lane = threadIdx.x & 31;
  int y  = wave & 15;
  int tb = wave >> 4;
  int n       = lane & 15;
  int cinbase = (lane >> 4) << 4;
  v8f c0 = {}, c1 = {}, c2 = {}, c3 = {};
  const _Float16* bptr = p2 + (size_t)tb * (P2_IMG * 32)
                            + ((size_t)y * P2_ROW + n) * 32 + cinbase;
#pragma unroll
  for (int dy = 0; dy < 3; ++dy)
#pragma unroll
    for (int dx = 0; dx < 3; ++dx) {
      int tap = dy * 3 + dx;
      v16h bfrag = *(const v16h*)(bptr + (dy * P2_ROW + dx) * 32);
      v16h a0 = *(const v16h*)(wf3 + ((tap * 4 + 0) * 32 + lane) * 16);
      v16h a1 = *(const v16h*)(wf3 + ((tap * 4 + 1) * 32 + lane) * 16);
      v16h a2 = *(const v16h*)(wf3 + ((tap * 4 + 2) * 32 + lane) * 16);
      v16h a3 = *(const v16h*)(wf3 + ((tap * 4 + 3) * 32 + lane) * 16);
      c0 = __builtin_amdgcn_wmma_f32_16x16x32_f16(false, a0, false, bfrag, (short)0, c0, false, false);
      c1 = __builtin_amdgcn_wmma_f32_16x16x32_f16(false, a1, false, bfrag, (short)0, c1, false, false);
      c2 = __builtin_amdgcn_wmma_f32_16x16x32_f16(false, a2, false, bfrag, (short)0, c2, false, false);
      c3 = __builtin_amdgcn_wmma_f32_16x16x32_f16(false, a3, false, bfrag, (short)0, c3, false, false);
    }
  int mAdd = (lane >> 4) * 8;
  size_t obase = ((size_t)tb * 256 + y * 16 + n) * 64;
#pragma unroll
  for (int r = 0; r < 8; ++r) {
    y3[obase + mAdd + r]      = c0[r];
    y3[obase + 16 + mAdd + r] = c1[r];
    y3[obase + 32 + mAdd + r] = c2[r];
    y3[obase + 48 + mAdd + r] = c3[r];
  }
}

// --------------------------- fused BN + LIF + pool + time-mean (block3 + output)
// y3 (tb,256,64) f32 -> out (b, c*64 + oy*8 + ox) f32.
__global__ void final_kernel(const float* __restrict__ y3, const float* __restrict__ ss,
                             float* __restrict__ out) {
  int tid  = blockIdx.x * blockDim.x + threadIdx.x;  // 64*64*64
  int c    = tid & 63;
  int opix = (tid >> 6) & 63;
  int b    = tid >> 12;
  int oy = opix >> 3, ox = opix & 7;
  float scale = ss[c], shift = ss[64 + c];
  float v0 = 0.f, v1 = 0.f, v2 = 0.f, v3 = 0.f, acc = 0.f;
#pragma unroll
  for (int t = 0; t < TT; ++t) {
    size_t base = ((size_t)(t * BB + b) * 256 + (2 * oy) * 16 + 2 * ox) * 64 + c;
    float x0 = fmaf(scale, y3[base],        shift);
    float x1 = fmaf(scale, y3[base + 64],   shift);
    float x2 = fmaf(scale, y3[base + 1024], shift);
    float x3 = fmaf(scale, y3[base + 1088], shift);
    v0 = 0.5f * (v0 + x0); v1 = 0.5f * (v1 + x1);
    v2 = 0.5f * (v2 + x2); v3 = 0.5f * (v3 + x3);
    float s0 = (v0 >= 1.f) ? 1.f : 0.f;  float s1 = (v1 >= 1.f) ? 1.f : 0.f;
    float s2 = (v2 >= 1.f) ? 1.f : 0.f;  float s3 = (v3 >= 1.f) ? 1.f : 0.f;
    v0 *= (1.f - s0); v1 *= (1.f - s1); v2 *= (1.f - s2); v3 *= (1.f - s3);
    acc += fmaxf(fmaxf(s0, s1), fmaxf(s2, s3));
  }
  out[b * 4096 + c * 64 + opix] = acc * 0.125f;
}

// ------------------------------------------------------------------------- launch
extern "C" void kernel_launch(void* const* d_in, const int* in_sizes, int n_in,
                              void* d_out, int out_size, void* d_ws, size_t ws_size,
                              hipStream_t stream) {
  const float* x   = (const float*)d_in[0];
  const float* w1  = (const float*)d_in[1];
  const float* g1  = (const float*)d_in[3];
  const float* be1 = (const float*)d_in[4];
  const float* w2  = (const float*)d_in[5];
  const float* g2  = (const float*)d_in[7];
  const float* be2 = (const float*)d_in[8];
  const float* w3  = (const float*)d_in[9];
  const float* g3  = (const float*)d_in[11];
  const float* be3 = (const float*)d_in[12];
  float* out = (float*)d_out;
  // conv biases b1/b2/b3 are absorbed by training-mode BN -> intentionally unused.

  char* ws = (char*)d_ws;
  size_t off = 0;
  float*    y1   = (float*)   (ws + off); off += 64u * 4096 * 32 * 4;         // 32 MB
  _Float16* p1   = (_Float16*)(ws + off); off += 512u * P1_IMG * 32 * 2;      // ~36 MB (padded)
  float*    y2   = (float*)   (ws + off); off += 512u * 1024 * 32 * 4;        // 64 MB
  _Float16* p2   = (_Float16*)(ws + off); off += 512u * P2_IMG * 32 * 2;      // ~10 MB (padded)
  float*    y3   = (float*)   (ws + off); off += 512u * 256 * 64 * 4;         // 32 MB
  _Float16* wf2  = (_Float16*)(ws + off); off += 32768;                       // 9*2*32*16 f16
  _Float16* wf3  = (_Float16*)(ws + off); off += 65536;                       // 9*4*32*16 f16
  float*    part = (float*)   (ws + off); off += 256u * 2 * 64 * 4;
  float*    ss1  = (float*)   (ws + off);
  float*    ss2  = ss1 + 64;
  float*    ss3  = ss2 + 64;

  // Zero halo-padded spike buffers (interiors rewritten by producers below).
  zero_kernel<<<2048, 256, 0, stream>>>((float4*)p1, (int)(512u * P1_IMG * 32 * 2 / 16));
  zero_kernel<<<1024, 256, 0, stream>>>((float4*)p2, (int)(512u * P2_IMG * 32 * 2 / 16));

  // Pack conv2/conv3 weights into WMMA A-fragment layout (f16).
  pack_w_kernel<<<36, 256, 0, stream>>>(w2, wf2, 2, 9 * 2 * 32 * 16);
  pack_w_kernel<<<72, 256, 0, stream>>>(w3, wf3, 4, 9 * 4 * 32 * 16);

  // Block 1 (conv once: input T-invariant).
  conv1_kernel<<<32768, 256, 0, stream>>>(x, w1, y1);
  stats_partial_kernel<<<256, 256, 0, stream>>>(y1, 32, 64 * 4096, part);
  finalize_stats_kernel<<<1, 64, 0, stream>>>(part, 256, 32, 1.f / (64.f * 4096.f), g1, be1, ss1);
  bn_lif_pool1_kernel<<<8192, 256, 0, stream>>>(y1, ss1, p1);

  // Block 2.
  conv2_wmma_kernel<<<4096, 256, 0, stream>>>(p1, wf2, y2);
  stats_partial_kernel<<<256, 256, 0, stream>>>(y2, 32, 512 * 1024, part);
  finalize_stats_kernel<<<1, 64, 0, stream>>>(part, 256, 32, 1.f / (512.f * 1024.f), g2, be2, ss2);
  bn_lif_pool2_kernel<<<2048, 256, 0, stream>>>(y2, ss2, p2);

  // Block 3 + output.
  conv3_wmma_kernel<<<1024, 256, 0, stream>>>(p2, wf3, y3);
  stats_partial_kernel<<<256, 256, 0, stream>>>(y3, 64, 512 * 256, part);
  finalize_stats_kernel<<<1, 64, 0, stream>>>(part, 256, 64, 1.f / (512.f * 256.f), g3, be3, ss3);
  final_kernel<<<1024, 256, 0, stream>>>(y3, ss3, out);
}
